// SelectiveSSM_86242943304172
// MI455X (gfx1250) — matrix operand order repaired and checked by reading, not compiled, not run on hardware
//
#include <hip/hip_runtime.h>
#include <hip/hip_bf16.h>

typedef __attribute__((ext_vector_type(16))) __bf16 v16bf;
typedef __attribute__((ext_vector_type(8)))  float  v8f;

#define WMMA_BF16(a, b, c) \
    __builtin_amdgcn_wmma_f32_16x16x32_bf16(false, (a), false, (b), (short)0, (c), false, false)

#define B_  8
#define T_  2048
#define D_  1024
#define N_  128
#define BT_ (B_ * T_)

// ---------------------------------------------------------------------------
// CDNA5 async global->LDS copy (ASYNCcnt-tracked, no VGPR roundtrip).
// lds_off = low 32 bits of the flat shared-memory pointer (LDS aperture rule).
// ---------------------------------------------------------------------------
__device__ __forceinline__ void async_ld_b128(void* lds_ptr, const void* gptr) {
    const unsigned lds_off = (unsigned)(unsigned long long)lds_ptr;
    asm volatile("global_load_async_to_lds_b128 %0, %1, off"
                 :: "v"(lds_off), "v"(gptr) : "memory");
}
__device__ __forceinline__ void wait_async0() {
    asm volatile("s_wait_asynccnt 0x0" ::: "memory");
}
__device__ __forceinline__ void wait_async2() {
    asm volatile("s_wait_asynccnt 0x2" ::: "memory");
}

// ---------------------------------------------------------------------------
// Swizzled-fragment load: each B-operand fragment stored as a contiguous
// 1 KB block (32 lanes x 16 bf16).  -> two coalesced global_load_b128 / lane.
// ---------------------------------------------------------------------------
__device__ __forceinline__ v16bf load_frag_sw(const __bf16* __restrict__ Wsw,
                                              int ntile, int KC, int kc, int lane) {
    const __bf16* p = Wsw + (((size_t)ntile * KC + kc) * 32 + lane) * 16;
    union { uint4 u[2]; v16bf v; } t;
    t.u[0] = ((const uint4*)p)[0];
    t.u[1] = ((const uint4*)p)[1];
    return t.v;
}

// ---------------------------------------------------------------------------
// Prologue 1: x (f32, row-major) -> bf16 row-major.  8 elems / thread.
// ---------------------------------------------------------------------------
__global__ __launch_bounds__(256) void k_cvt_x(const float* __restrict__ x,
                                               __bf16* __restrict__ xb) {
    const size_t i = ((size_t)blockIdx.x * 256 + threadIdx.x) * 8;
    const float4 a = ((const float4*)(x + i))[0];
    const float4 b = ((const float4*)(x + i))[1];
    union { uint4 u; __bf16 h[8]; } t;
    t.h[0] = (__bf16)a.x; t.h[1] = (__bf16)a.y;
    t.h[2] = (__bf16)a.z; t.h[3] = (__bf16)a.w;
    t.h[4] = (__bf16)b.x; t.h[5] = (__bf16)b.y;
    t.h[6] = (__bf16)b.z; t.h[7] = (__bf16)b.w;
    *((uint4*)(xb + i)) = t.u;
}

// ---------------------------------------------------------------------------
// Prologue 2: weight (rows x K f32, row-major; used as @W^T) -> bf16
// fragment-major layout.  One lane-fragment (16 elems) per thread.
// ---------------------------------------------------------------------------
__global__ __launch_bounds__(256) void k_swz(const float* __restrict__ W,
                                             __bf16* __restrict__ out, int K, int KC) {
    const int gid   = blockIdx.x * 256 + threadIdx.x;
    const int lane  = gid & 31;
    const int frag  = gid >> 5;
    const int kc    = frag % KC;
    const int ntile = frag / KC;
    const float* p  = W + (size_t)(ntile * 16 + (lane & 15)) * K + kc * 32 + 8 * (lane >> 4);
    __bf16* o       = out + (size_t)gid * 16;
#pragma unroll
    for (int e = 0; e < 16; ++e) o[e] = (__bf16)p[(e & 7) + 16 * (e >> 3)];
}

// ---------------------------------------------------------------------------
// Kernel 1: u[bt, n] = sum_d x[bt, d] * Bw[n, d]     (16384 x 128 x K=1024)
// One block per 16-row x tile (async-staged to LDS); wave w -> N-tile w.
// ---------------------------------------------------------------------------
__global__ __launch_bounds__(256) void k_u(const __bf16* __restrict__ xb,
                                           const __bf16* __restrict__ Bsw,
                                           float* __restrict__ u) {
    __shared__ alignas(16) __bf16 xs[16][D_ + 8];
    const int tid  = threadIdx.x;
    const int lane = tid & 31;
    const int wave = tid >> 5;            // 0..7 -> n-tile
    const int row0 = blockIdx.x * 16;

#pragma unroll
    for (int it = 0; it < 8; ++it) {      // 16 rows x 128 b128-chunks
        const int idx = tid + it * 256;
        const int r = idx >> 7, c = idx & 127;
        async_ld_b128(&xs[r][c * 8], xb + (size_t)(row0 + r) * D_ + c * 8);
    }
    wait_async0();
    __syncthreads();

    const int m    = lane & 15;
    const int half = lane >> 4;
    const int n0   = wave * 16;

    v8f acc = {};
    for (int kc = 0; kc < D_ / 32; ++kc) {
        const int k0 = kc * 32;
        v16bf a;
#pragma unroll
        for (int e = 0; e < 8; ++e) a[e]     = xs[m][k0 + 8 * half + e];
#pragma unroll
        for (int e = 0; e < 8; ++e) a[8 + e] = xs[m][k0 + 16 + 8 * half + e];
        const v16bf b = load_frag_sw(Bsw, wave, D_ / 32, kc, lane);
        acc = WMMA_BF16(a, b, acc);
    }
#pragma unroll
    for (int r = 0; r < 8; ++r)
        u[(size_t)(row0 + r + 8 * half) * N_ + n0 + m] = acc[r];
}

// ---------------------------------------------------------------------------
// Kernel 2: sequential scan  h_t = h_{t-1} @ A + u_t,  hs[b,t,:] = h_t
// Single block, 8 waves; h (16x128, rows 8..15 stay 0) lives in LDS as bf16.
// ---------------------------------------------------------------------------
__global__ __launch_bounds__(256) void k_scan(const float* __restrict__ Amat,
                                              const float* __restrict__ u,
                                              __bf16* __restrict__ hs) {
    __shared__ alignas(16) __bf16 h[16][N_ + 8];
    const int tid  = threadIdx.x;
    const int lane = tid & 31;
    const int wave = tid >> 5;
    const int m    = lane & 15;
    const int half = lane >> 4;
    const int n0   = wave * 16;

    for (int idx = tid; idx < 16 * (N_ + 8); idx += 256)
        (&h[0][0])[idx] = (__bf16)0.0f;

    // Preload the 4 B-operand fragments of A: element (k, n) = Amat[k*128 + n]
    v16bf bfr[4];
#pragma unroll
    for (int kc = 0; kc < 4; ++kc) {
#pragma unroll
        for (int e = 0; e < 16; ++e)
            bfr[kc][e] = (__bf16)Amat[(size_t)(kc * 32 + 8 * half + (e & 7) + 16 * (e >> 3)) * N_ + n0 + m];
    }
    __syncthreads();

    for (int t = 0; t < T_; ++t) {
        v8f acc;
        if (half == 0) {
#pragma unroll
            for (int r = 0; r < 8; ++r)        // rows 0..7 = batches
                acc[r] = u[((size_t)r * T_ + t) * N_ + n0 + m];
        } else {
#pragma unroll
            for (int r = 0; r < 8; ++r) acc[r] = 0.0f;  // padded rows
        }

#pragma unroll
        for (int kc = 0; kc < 4; ++kc) {
            const int k0 = kc * 32;
            v16bf a;
#pragma unroll
            for (int e = 0; e < 8; ++e) a[e]     = h[m][k0 + 8 * half + e];
#pragma unroll
            for (int e = 0; e < 8; ++e) a[8 + e] = h[m][k0 + 16 + 8 * half + e];
            acc = WMMA_BF16(a, bfr[kc], acc);
        }

        __syncthreads();                       // everyone done reading old h
#pragma unroll
        for (int r = 0; r < 8; ++r)
            h[r + 8 * half][n0 + m] = (__bf16)acc[r];
        if (half == 0) {
#pragma unroll
            for (int r = 0; r < 8; ++r)
                hs[((size_t)r * T_ + t) * N_ + n0 + m] = (__bf16)acc[r];
        }
        __syncthreads();                       // new h visible
    }
}

// ---------------------------------------------------------------------------
// Kernel 3: y = hs @ C^T + x @ Dw^T + Db
// Block tile: 32 (M) x 512 (D-cols); grid = (BT/32, 2).
// Double-buffered async x panels overlap the next fetch with WMMA work.
// ---------------------------------------------------------------------------
__global__ __launch_bounds__(256) void k_y(const __bf16* __restrict__ xb,
                                           const __bf16* __restrict__ hs,
                                           const __bf16* __restrict__ Csw,
                                           const __bf16* __restrict__ Dsw,
                                           const float* __restrict__ Db,
                                           float* __restrict__ y) {
    __shared__ alignas(16) __bf16 xs[2][32][128 + 8];
    __shared__ alignas(16) __bf16 hss[32][N_ + 8];
    const int tid   = threadIdx.x;
    const int lane  = tid & 31;
    const int wave  = tid >> 5;
    const int row0  = blockIdx.x * 32;
    const int m     = lane & 15;
    const int half  = lane >> 4;
    const int dbase = blockIdx.y * 512 + wave * 64;
    const int nt0   = dbase >> 4;             // first d-tile index of this wave

    // per-thread staging coords: 32 rows x 16 b128-chunks, 2 chunks / thread
    const int r0 = tid >> 4,         c0 = tid & 15;         // chunk tid
    const int r1 = (tid + 256) >> 4, c1 = (tid + 256) & 15; // chunk tid+256

    // hs panel + x panel 0 (async, in-order completion)
    async_ld_b128(&hss[r0][c0 * 8], hs + (size_t)(row0 + r0) * N_ + c0 * 8);
    async_ld_b128(&hss[r1][c1 * 8], hs + (size_t)(row0 + r1) * N_ + c1 * 8);
    async_ld_b128(&xs[0][r0][c0 * 8], xb + (size_t)(row0 + r0) * D_ + c0 * 8);
    async_ld_b128(&xs[0][r1][c1 * 8], xb + (size_t)(row0 + r1) * D_ + c1 * 8);

    v8f acc[2][4];
#pragma unroll
    for (int mt = 0; mt < 2; ++mt)
#pragma unroll
        for (int j = 0; j < 4; ++j) acc[mt][j] = (v8f){};

    for (int kg = 0; kg < 8; ++kg) {          // K panels of 128 (total K=1024)
        __syncthreads();                      // next buffer fully consumed
        if (kg < 7) {                         // prefetch panel kg+1 async
            const int nb = (kg + 1) & 1;
            const size_t koff = (size_t)(kg + 1) * 128;
            async_ld_b128(&xs[nb][r0][c0 * 8], xb + (size_t)(row0 + r0) * D_ + koff + c0 * 8);
            async_ld_b128(&xs[nb][r1][c1 * 8], xb + (size_t)(row0 + r1) * D_ + koff + c1 * 8);
            wait_async2();                    // panel kg (and hss) now resident
        } else {
            wait_async0();
        }
        __syncthreads();

        const int cb = kg & 1;
#pragma unroll
        for (int kc = 0; kc < 4; ++kc) {
            const int kl = kc * 32;           // K offset inside panel
            v16bf a0, a1;
#pragma unroll
            for (int e = 0; e < 8; ++e) {
                a0[e]     = xs[cb][m][kl + 8 * half + e];
                a0[8 + e] = xs[cb][m][kl + 16 + 8 * half + e];
                a1[e]     = xs[cb][16 + m][kl + 8 * half + e];
                a1[8 + e] = xs[cb][16 + m][kl + 16 + 8 * half + e];
            }
#pragma unroll
            for (int j = 0; j < 4; ++j) {
                const v16bf b = load_frag_sw(Dsw, nt0 + j, D_ / 32, kg * 4 + kc, lane);
                acc[0][j] = WMMA_BF16(a0, b, acc[0][j]);
                acc[1][j] = WMMA_BF16(a1, b, acc[1][j]);
            }
        }
    }

    // hs @ C^T contribution (K = 128)
#pragma unroll
    for (int kc = 0; kc < 4; ++kc) {
        const int k0 = kc * 32;
        v16bf a0, a1;
#pragma unroll
        for (int e = 0; e < 8; ++e) {
            a0[e]     = hss[m][k0 + 8 * half + e];
            a0[8 + e] = hss[m][k0 + 16 + 8 * half + e];
            a1[e]     = hss[16 + m][k0 + 8 * half + e];
            a1[8 + e] = hss[16 + m][k0 + 16 + 8 * half + e];
        }
#pragma unroll
        for (int j = 0; j < 4; ++j) {
            const v16bf b = load_frag_sw(Csw, nt0 + j, N_ / 32, kc, lane);
            acc[0][j] = WMMA_BF16(a0, b, acc[0][j]);
            acc[1][j] = WMMA_BF16(a1, b, acc[1][j]);
        }
    }

    // bias + store
#pragma unroll
    for (int j = 0; j < 4; ++j) {
        const int d    = dbase + j * 16 + m;
        const float bv = Db[d];
#pragma unroll
        for (int mt = 0; mt < 2; ++mt) {
#pragma unroll
            for (int r = 0; r < 8; ++r)
                y[(size_t)(row0 + mt * 16 + r + 8 * half) * D_ + d] = acc[mt][j][r] + bv;
        }
    }
}

// ---------------------------------------------------------------------------
extern "C" void kernel_launch(void* const* d_in, const int* in_sizes, int n_in,
                              void* d_out, int out_size, void* d_ws, size_t ws_size,
                              hipStream_t stream) {
    const float* x    = (const float*)d_in[0];
    const float* Amat = (const float*)d_in[1];   // (1,128,128)
    const float* Bw   = (const float*)d_in[2];   // (1,128,1024)
    const float* Cw   = (const float*)d_in[3];   // (1,1024,128)
    const float* Dw   = (const float*)d_in[4];   // (1024,1024)
    const float* Db   = (const float*)d_in[5];   // (1024,)
    float* y          = (float*)d_out;

    char* ws = (char*)d_ws;
    float*  u    = (float*)ws;                                   ws += (size_t)BT_ * N_ * 4;  //  8 MB
    __bf16* hsb  = (__bf16*)ws;                                  ws += (size_t)BT_ * N_ * 2;  //  4 MB
    __bf16* xbf  = (__bf16*)ws;                                  ws += (size_t)BT_ * D_ * 2;  // 32 MB
    __bf16* Bsw  = (__bf16*)ws;                                  ws += (size_t)N_ * D_ * 2;   // 256 KB
    __bf16* Dsw  = (__bf16*)ws;                                  ws += (size_t)D_ * D_ * 2;   //  2 MB
    __bf16* Csw  = (__bf16*)ws;                                  //  256 KB

    // Prologue: one-time bf16 conversion / fragment swizzle
    k_cvt_x<<<(BT_ * (D_ / 8)) / 256, 256, 0, stream>>>(x, xbf);
    k_swz  <<<(  (N_/16) * (D_/32) * 32) / 256, 256, 0, stream>>>(Bw, Bsw, D_, D_ / 32);
    k_swz  <<<(  (D_/16) * (D_/32) * 32) / 256, 256, 0, stream>>>(Dw, Dsw, D_, D_ / 32);
    k_swz  <<<(  (D_/16) * (N_/32) * 32) / 256, 256, 0, stream>>>(Cw, Csw, N_, N_ / 32);

    // Main pipeline
    k_u   <<<BT_ / 16, 256, 0, stream>>>(xbf, Bsw, u);
    k_scan<<<1,        256, 0, stream>>>(Amat, u, hsb);
    k_y   <<<dim3(BT_ / 32, 2), 256, 0, stream>>>(xbf, hsb, Csw, Dsw, Db, y);
}
